// CAMixer_5617817223396
// MI455X (gfx1250) — matrix operand order repaired and checked by reading, not compiled, not run on hardware
//
#include <hip/hip_runtime.h>
#include <hip/hip_bf16.h>
#include <math.h>

// ---------------- problem constants ----------------
#define DIMC   64
#define HH     256
#define WW     256
#define HW     65536          // H*W
#define BATCH  4
#define WSZ    8
#define NWIN   1024           // (H/WS)*(W/WS) per batch
#define C4N    17             // (DIM+4)/4
#define C8N    8              // (DIM+4)/8
#define CDIMN  68

typedef __attribute__((ext_vector_type(16))) _Float16 v16h;
typedef __attribute__((ext_vector_type(8)))  float    v8f;

// ---- WMMA 16-bit fragment K-index maps (per CDNA5 ISA 7.12.2) ----
// A matrix 16x32: lanes 0-15 row M=lane, VGPR0-3 K=0..7, VGPR4-7 K=16..23;
//                 lanes 16-31 same rows, K blocks 8..15 / 24..31.
__device__ __forceinline__ int a_kof(int j, int half) {
  return (j < 4) ? (half * 8 + 2 * j) : (16 + half * 8 + 2 * (j - 4));
}
// B matrix 32x16: lanes 0-15 col N=lane, VGPR0-3 K=0..7, VGPR4-7 K=16..23;
//                 lanes 16-31 same cols, K blocks 8..15 / 24..31.
__device__ __forceinline__ int b_kof(int j, int half) {
  return (j < 4) ? (half * 8 + 2 * j) : (16 + half * 8 + 2 * (j - 4));
}

__device__ __forceinline__ float leakyf(float x) { return x > 0.f ? x : 0.1f * x; }
__device__ __forceinline__ float sigmoidf_(float x) { return 1.f / (1.f + __expf(-x)); }

// =================================================================
// 1x1 conv (64->64) as GEMM: out[b,oc,hw] = W[oc,:]·in[b,:,hw]+bias.
// Block = 8 waves = 128 pixels. Input tile + weights staged in LDS via
// coalesced float4 loads; each wave computes 16 px x 64 oc (8 WMMA).
// =================================================================
__global__ __launch_bounds__(256)
void conv1x1_wmma(const float* __restrict__ in, const float* __restrict__ wgt,
                  const float* __restrict__ bias, float* __restrict__ out) {
  __shared__ _Float16 Lw[64 * 64];    // weights [oc][ic]   8KB
  __shared__ _Float16 Li[64 * 128];   // inputs  [ic][p]   16KB
  const int tid = threadIdx.x;
  const int lane = tid & 31, wave = tid >> 5;
  const int lr = lane & 15, half = lane >> 4;
  const long p0 = (long)blockIdx.x * 128;     // block pixel base
  const long bb = p0 >> 16;                   // / HW
  const long hw = p0 & (HW - 1);
  const float* inb = in + bb * DIMC * (long)HW + hw;
  float* outb = out + bb * DIMC * (long)HW + hw + wave * 16;

  // stage weights as f16
  for (int i = tid; i < 4096; i += 256) Lw[i] = (_Float16)wgt[i];
  // stage input tile, coalesced 16B loads
  for (int i = tid; i < 2048; i += 256) {
    const int c = i >> 5, seg = i & 31;
    const float4 f = *(const float4*)(inb + (long)c * HW + seg * 4);
    Li[c * 128 + seg * 4 + 0] = (_Float16)f.x;
    Li[c * 128 + seg * 4 + 1] = (_Float16)f.y;
    Li[c * 128 + seg * 4 + 2] = (_Float16)f.z;
    Li[c * 128 + seg * 4 + 3] = (_Float16)f.w;
  }
  __syncthreads();

  // B fragments: K = input channel, N = pixel (this wave's 16 columns)
  v16h bf[2];
#pragma unroll
  for (int kk = 0; kk < 2; ++kk) {
#pragma unroll
    for (int j = 0; j < 8; ++j) {
      const int k0 = kk * 32 + b_kof(j, half);
      bf[kk][2 * j]     = Li[k0 * 128 + wave * 16 + lr];
      bf[kk][2 * j + 1] = Li[(k0 + 1) * 128 + wave * 16 + lr];
    }
  }
#pragma unroll
  for (int mt = 0; mt < 4; ++mt) {
    v8f acc = {};
#pragma unroll
    for (int kk = 0; kk < 2; ++kk) {
      v16h af;
      const int oc = mt * 16 + lr;
#pragma unroll
      for (int j = 0; j < 8; ++j) {
        const int k0 = kk * 32 + a_kof(j, half);
        af[2 * j]     = Lw[oc * 64 + k0];
        af[2 * j + 1] = Lw[oc * 64 + k0 + 1];
      }
      acc = __builtin_amdgcn_wmma_f32_16x16x32_f16(false, af, false, bf[kk],
                                                   (short)0, acc, false, false);
    }
#pragma unroll
    for (int r = 0; r < 8; ++r) {
      const int oc = mt * 16 + r + 8 * half;
      outb[(long)oc * HW + lr] = acc[r] + bias[oc];
    }
  }
}

// =================================================================
// Predictor: cond = [v, cond_global, cond_wind]; t = leaky(LN(in_w@cond));
// offsets = tanh(off2(leaky(off1(t))))*8 ; tmean ; per-batch t sums.
// =================================================================
__global__ __launch_bounds__(256)
void predictor_kernel(const float* __restrict__ v, const float* __restrict__ cg,
                      const float* __restrict__ in_w, const float* __restrict__ in_b,
                      const float* __restrict__ ln_w, const float* __restrict__ ln_b,
                      const float* __restrict__ o1_w, const float* __restrict__ o1_b,
                      const float* __restrict__ o2_w, const float* __restrict__ o2_b,
                      float* __restrict__ t, float* __restrict__ tmean,
                      float* __restrict__ offs, float* __restrict__ tsum) {
  __shared__ float sw[1369];
  __shared__ float tacc[C4N];
  const int tid = threadIdx.x;
  for (int i = tid; i < 1369; i += 256) {
    float val;
    if (i < 1156)      val = in_w[i];
    else if (i < 1173) val = in_b[i - 1156];
    else if (i < 1190) val = ln_w[i - 1173];
    else if (i < 1207) val = ln_b[i - 1190];
    else if (i < 1343) val = o1_w[i - 1207];
    else if (i < 1351) val = o1_b[i - 1343];
    else if (i < 1367) val = o2_w[i - 1351];
    else               val = o2_b[i - 1367];
    sw[i] = val;
  }
  if (tid < C4N) tacc[tid] = 0.f;
  __syncthreads();
  const float* s_inw = sw;          const float* s_inb = sw + 1156;
  const float* s_lnw = sw + 1173;   const float* s_lnb = sw + 1190;
  const float* s_o1w = sw + 1207;   const float* s_o1b = sw + 1343;
  const float* s_o2w = sw + 1351;   const float* s_o2b = sw + 1367;

  const long idx = (long)blockIdx.x * 256 + tid;   // pixel id, < B*HW
  const long bb = idx >> 16;
  const long hw = idx & (HW - 1);
  const int h = (int)(hw >> 8), wc = (int)(hw & 255);

  float cond[CDIMN];
#pragma unroll
  for (int c = 0; c < DIMC; ++c) cond[c] = v[(bb * DIMC + c) * (long)HW + hw];
  cond[64] = cg[(bb * 2 + 0) * (long)HW + hw];
  cond[65] = cg[(bb * 2 + 1) * (long)HW + hw];
  cond[66] = -1.f + (2.f / 7.f) * (float)(h & 7);   // wy
  cond[67] = -1.f + (2.f / 7.f) * (float)(wc & 7);  // wx

  float pre[C4N];
  float u = 0.f;
#pragma unroll
  for (int c = 0; c < C4N; ++c) {
    float a = s_inb[c];
    for (int k = 0; k < CDIMN; ++k) a += s_inw[c * CDIMN + k] * cond[k];
    pre[c] = a; u += a;
  }
  u *= (1.f / C4N);
  float s2 = 0.f;
#pragma unroll
  for (int c = 0; c < C4N; ++c) { float d = pre[c] - u; s2 += d * d; }
  s2 *= (1.f / C4N);
  const float rinv = rsqrtf(s2 + 1e-6f);
  float tm = 0.f;
  float tv[C4N];
#pragma unroll
  for (int c = 0; c < C4N; ++c) {
    float xn = (pre[c] - u) * rinv;
    float tc = leakyf(s_lnw[c] * xn + s_lnb[c]);
    tv[c] = tc; tm += tc;
    t[(bb * C4N + c) * (long)HW + hw] = tc;
    atomicAdd(&tacc[c], tc);
  }
  tmean[idx] = tm * (1.f / C4N);

  float h1[C8N];
#pragma unroll
  for (int j = 0; j < C8N; ++j) {
    float a = s_o1b[j];
    for (int c = 0; c < C4N; ++c) a += s_o1w[j * C4N + c] * tv[c];
    h1[j] = leakyf(a);
  }
#pragma unroll
  for (int d = 0; d < 2; ++d) {
    float a = s_o2b[d];
    for (int j = 0; j < C8N; ++j) a += s_o2w[d * C8N + j] * h1[j];
    offs[(bb * 2 + d) * (long)HW + hw] = tanhf(a) * 8.f;
  }
  __syncthreads();
  if (tid < C4N) atomicAdd(&tsum[bb * C4N + tid], tacc[tid]);
}

// ================= sa: 3x3 conv (17ch -> 1) + sigmoid =================
__global__ __launch_bounds__(256)
void sa_kernel(const float* __restrict__ t, const float* __restrict__ sa_w,
               const float* __restrict__ sa_b, float* __restrict__ sa) {
  const long idx = (long)blockIdx.x * 256 + threadIdx.x;
  const long bb = idx >> 16;
  const long hw = idx & (HW - 1);
  const int h = (int)(hw >> 8), wc = (int)(hw & 255);
  float acc = sa_b[0];
  for (int ky = -1; ky <= 1; ++ky) {
    int yy = h + ky; if (yy < 0 || yy >= HH) continue;
    for (int kx = -1; kx <= 1; ++kx) {
      int xx = wc + kx; if (xx < 0 || xx >= WW) continue;
      const long p = (long)yy * WW + xx;
      for (int c = 0; c < C4N; ++c)
        acc += sa_w[c * 9 + (ky + 1) * 3 + (kx + 1)] * t[(bb * C4N + c) * (long)HW + p];
    }
  }
  sa[idx] = sigmoidf_(acc);
}

// ================= ca: sigmoid(ca_w @ mean_t + ca_b) per batch =================
__global__ void ca_kernel(const float* __restrict__ tsum, const float* __restrict__ ca_w,
                          const float* __restrict__ ca_b, float* __restrict__ ca) {
  const int bb = blockIdx.x, oc = threadIdx.x;
  float acc = ca_b[oc];
  for (int c = 0; c < C4N; ++c)
    acc += ca_w[oc * C4N + c] * tsum[bb * C4N + c] * (1.f / (float)HW);
  ca[bb * DIMC + oc] = sigmoidf_(acc);
}

// ================= window mask: MLP + gumbel-hard =================
__global__ __launch_bounds__(256)
void mask_kernel(const float* __restrict__ tmean, const float* __restrict__ gu,
                 const float* __restrict__ m1_w, const float* __restrict__ m1_b,
                 const float* __restrict__ m2_w, const float* __restrict__ m2_b,
                 float* __restrict__ mask) {
  const int wi = blockIdx.x * 256 + threadIdx.x;   // < B*NWIN
  const int bb = wi >> 10, win = wi & 1023;
  const int wh = win >> 5, wwc = win & 31;
  float xm[64];
#pragma unroll
  for (int p = 0; p < 64; ++p)
    xm[p] = tmean[(long)bb * HW + (long)(wh * 8 + (p >> 3)) * WW + (wwc * 8 + (p & 7))];
  float h1[8];
#pragma unroll
  for (int j = 0; j < 8; ++j) {
    float a = m1_b[j];
    for (int p = 0; p < 64; ++p) a += m1_w[j * 64 + p] * xm[p];
    h1[j] = leakyf(a);
  }
  float lg[2];
#pragma unroll
  for (int d = 0; d < 2; ++d) {
    float a = m2_b[d];
    for (int j = 0; j < 8; ++j) a += m2_w[d * 8 + j] * h1[j];
    lg[d] = a;
  }
  const float mx = fmaxf(lg[0], lg[1]);
  const float e0 = __expf(lg[0] - mx), e1 = __expf(lg[1] - mx);
  const float p0 = e0 / (e0 + e1), p1 = e1 / (e0 + e1);
  const float u0 = gu[(long)wi * 2 + 0], u1 = gu[(long)wi * 2 + 1];
  const float g0 = -logf(-logf(u0 + 1e-10f) + 1e-10f);
  const float g1 = -logf(-logf(u1 + 1e-10f) + 1e-10f);
  mask[wi] = ((p0 + g0) >= (p1 + g1)) ? 1.f : 0.f;
}

// ================= k_img = x + flow_warp(x, offsets) =================
__global__ __launch_bounds__(256)
void warp_kernel(const float* __restrict__ x, const float* __restrict__ offs,
                 float* __restrict__ kimg) {
  const long idx = (long)blockIdx.x * 256 + threadIdx.x;
  const long bb = idx >> 16;
  const long hw = idx & (HW - 1);
  const int h = (int)(hw >> 8), wc = (int)(hw & 255);
  const float ox = offs[(bb * 2 + 0) * (long)HW + hw];
  const float oy = offs[(bb * 2 + 1) * (long)HW + hw];
  const float px = fminf(fmaxf((float)wc + ox, 0.f), (float)(WW - 1));
  const float py = fminf(fmaxf((float)h + oy, 0.f), (float)(HH - 1));
  const float x0 = floorf(px), y0 = floorf(py);
  const float fx = px - x0, fy = py - y0;
  const int x0i = (int)x0, y0i = (int)y0;
  const int x1i = min(x0i + 1, WW - 1), y1i = min(y0i + 1, HH - 1);
  for (int c = 0; c < DIMC; ++c) {
    const float* base = x + (bb * DIMC + c) * (long)HW;
    const float v00 = base[(long)y0i * WW + x0i];
    const float v01 = base[(long)y0i * WW + x1i];
    const float v10 = base[(long)y1i * WW + x0i];
    const float v11 = base[(long)y1i * WW + x1i];
    const float bl = (1.f - fy) * ((1.f - fx) * v00 + fx * v01) +
                     fy * ((1.f - fx) * v10 + fx * v11);
    kimg[(bb * DIMC + c) * (long)HW + hw] = base[hw] + bl;
  }
}

// =================================================================
// Per-window attention: q=(x·m)@pq^T+b, k=(kimg·m)@pk^T+b,
// attn=softmax(q@k^T), out = attn@(v·m) + v·sa·(1-m).  8 waves/window.
// =================================================================
__global__ __launch_bounds__(256)
void attn_wmma(const float* __restrict__ x, const float* __restrict__ kimg,
               const float* __restrict__ v, const float* __restrict__ sa,
               const float* __restrict__ maskp,
               const float* __restrict__ pq_w, const float* __restrict__ pq_b,
               const float* __restrict__ pk_w, const float* __restrict__ pk_b,
               float* __restrict__ outsp) {
  __shared__ __align__(16) unsigned char smem[49152];
  _Float16* Xs  = (_Float16*)(smem);          // [p][c] masked x       (8KB)
  _Float16* Ksn = (_Float16*)(smem + 8192);   // [p][c] masked kimg    (8KB)
  float*    Sl  = (float*)(smem);             // [pi][pj] logits, overlays Xs+Ksn (16KB)
  _Float16* Vt  = (_Float16*)(smem + 16384);  // [c][p] raw v          (8KB)
  _Float16* Qs  = (_Float16*)(smem + 24576);  // [p][c] q1             (8KB)
  _Float16* Kp  = (_Float16*)(smem + 32768);  // [p][c] k1             (8KB)
  _Float16* Ps  = (_Float16*)(smem + 40960);  // [pi][pj] softmax      (8KB)

  const int tid = threadIdx.x;
  const int lane = tid & 31, wave = tid >> 5;
  const int lr = lane & 15, half = lane >> 4;
  const int wIdx = blockIdx.x;
  const long bb = wIdx >> 10;
  const int win = wIdx & 1023;
  const int row0 = (win >> 5) * WSZ, col0 = (win & 31) * WSZ;
  const float mf = maskp[wIdx];

  // ---- stage window (16B vectorized; window rows are 8 contiguous floats) ----
  for (int i = tid; i < 1024; i += 256) {
    const int c = i >> 4, rem = i & 15;
    const int row = rem >> 1, col4 = (rem & 1) * 4;
    const long g = (bb * DIMC + c) * (long)HW + (long)(row0 + row) * WW + (col0 + col4);
    const float4 fx = *(const float4*)(x + g);
    const float4 fk = *(const float4*)(kimg + g);
    const float4 fv = *(const float4*)(v + g);
    const int p = row * 8 + col4;
    Xs[(p + 0) * 64 + c] = (_Float16)(fx.x * mf);
    Xs[(p + 1) * 64 + c] = (_Float16)(fx.y * mf);
    Xs[(p + 2) * 64 + c] = (_Float16)(fx.z * mf);
    Xs[(p + 3) * 64 + c] = (_Float16)(fx.w * mf);
    Ksn[(p + 0) * 64 + c] = (_Float16)(fk.x * mf);
    Ksn[(p + 1) * 64 + c] = (_Float16)(fk.y * mf);
    Ksn[(p + 2) * 64 + c] = (_Float16)(fk.z * mf);
    Ksn[(p + 3) * 64 + c] = (_Float16)(fk.w * mf);
    Vt[c * 64 + p + 0] = (_Float16)fv.x;
    Vt[c * 64 + p + 1] = (_Float16)fv.y;
    Vt[c * 64 + p + 2] = (_Float16)fv.z;
    Vt[c * 64 + p + 3] = (_Float16)fv.w;
  }
  __syncthreads();

  // ---- q / k projections: 16 tiles each, 2 per wave ----
#pragma unroll
  for (int s = 0; s < 2; ++s) {
    const int tt = wave * 2 + s;
    const int mi = tt >> 2, nj = tt & 3;      // pixel tile, out-channel tile
    v8f aq = {}, ak = {};
#pragma unroll
    for (int kk = 0; kk < 2; ++kk) {
      v16h Aq, Ak, Bq, Bk;
      const int p = mi * 16 + lr;
      const int oc = nj * 16 + lr;
#pragma unroll
      for (int j = 0; j < 8; ++j) {
        const int ka = kk * 32 + a_kof(j, half);
        Aq[2 * j]     = Xs[p * 64 + ka];
        Aq[2 * j + 1] = Xs[p * 64 + ka + 1];
        Ak[2 * j]     = Ksn[p * 64 + ka];
        Ak[2 * j + 1] = Ksn[p * 64 + ka + 1];
        const int kb = kk * 32 + b_kof(j, half);
        Bq[2 * j]     = (_Float16)pq_w[oc * 64 + kb];     // B[ic][oc]
        Bq[2 * j + 1] = (_Float16)pq_w[oc * 64 + kb + 1];
        Bk[2 * j]     = (_Float16)pk_w[oc * 64 + kb];
        Bk[2 * j + 1] = (_Float16)pk_w[oc * 64 + kb + 1];
      }
      aq = __builtin_amdgcn_wmma_f32_16x16x32_f16(false, Aq, false, Bq, (short)0, aq, false, false);
      ak = __builtin_amdgcn_wmma_f32_16x16x32_f16(false, Ak, false, Bk, (short)0, ak, false, false);
    }
#pragma unroll
    for (int r = 0; r < 8; ++r) {
      const int p = mi * 16 + r + 8 * half;
      const int oc = nj * 16 + lr;
      Qs[p * 64 + oc] = (_Float16)(aq[r] + pq_b[oc]);
      Kp[p * 64 + oc] = (_Float16)(ak[r] + pk_b[oc]);
    }
  }
  __syncthreads();

  // ---- logits = q @ k^T ----
#pragma unroll
  for (int s = 0; s < 2; ++s) {
    const int tt = wave * 2 + s;
    const int mi = tt >> 2, nj = tt & 3;      // pi tile, pj tile
    v8f acc = {};
#pragma unroll
    for (int kk = 0; kk < 2; ++kk) {
      v16h A, Bm;
      const int pi = mi * 16 + lr;
      const int pj = nj * 16 + lr;
#pragma unroll
      for (int j = 0; j < 8; ++j) {
        const int ka = kk * 32 + a_kof(j, half);
        A[2 * j]     = Qs[pi * 64 + ka];
        A[2 * j + 1] = Qs[pi * 64 + ka + 1];
        const int kb = kk * 32 + b_kof(j, half);
        Bm[2 * j]     = Kp[pj * 64 + kb];                 // B[c][pj] = k1[pj][c]
        Bm[2 * j + 1] = Kp[pj * 64 + kb + 1];
      }
      acc = __builtin_amdgcn_wmma_f32_16x16x32_f16(false, A, false, Bm, (short)0, acc, false, false);
    }
#pragma unroll
    for (int r = 0; r < 8; ++r)
      Sl[(mi * 16 + r + 8 * half) * 64 + nj * 16 + lr] = acc[r];
  }
  __syncthreads();

  // ---- row softmax: 4 lanes per row, shfl_xor reductions, no arrays ----
  {
    const int row = tid >> 2, qd = tid & 3;
    const int jb = qd * 16;
    float mx = -3.4e38f;
#pragma unroll
    for (int j = 0; j < 16; ++j) mx = fmaxf(mx, Sl[row * 64 + jb + j]);
    mx = fmaxf(mx, __shfl_xor(mx, 1, 4));
    mx = fmaxf(mx, __shfl_xor(mx, 2, 4));
    float sum = 0.f;
#pragma unroll
    for (int j = 0; j < 16; ++j) sum += __expf(Sl[row * 64 + jb + j] - mx);
    sum += __shfl_xor(sum, 1, 4);
    sum += __shfl_xor(sum, 2, 4);
    const float inv = 1.f / sum;
#pragma unroll
    for (int j = 0; j < 16; ++j)
      Ps[row * 64 + jb + j] = (_Float16)(__expf(Sl[row * 64 + jb + j] - mx) * inv);
  }
  __syncthreads();

  // ---- out = attn @ (v·m)  + v·sa·(1-m) ----
#pragma unroll
  for (int s = 0; s < 2; ++s) {
    const int tt = wave * 2 + s;
    const int mi = tt >> 2, nj = tt & 3;      // pi tile, channel tile
    v8f acc = {};
#pragma unroll
    for (int kk = 0; kk < 2; ++kk) {
      v16h A, Bm;
      const int pi = mi * 16 + lr;
      const int c = nj * 16 + lr;
#pragma unroll
      for (int j = 0; j < 8; ++j) {
        const int ka = kk * 32 + a_kof(j, half);
        A[2 * j]     = Ps[pi * 64 + ka];
        A[2 * j + 1] = Ps[pi * 64 + ka + 1];
        const int kb = kk * 32 + b_kof(j, half);
        Bm[2 * j]     = (_Float16)((float)Vt[c * 64 + kb] * mf);   // B[pj][c] = v1
        Bm[2 * j + 1] = (_Float16)((float)Vt[c * 64 + kb + 1] * mf);
      }
      acc = __builtin_amdgcn_wmma_f32_16x16x32_f16(false, A, false, Bm, (short)0, acc, false, false);
    }
#pragma unroll
    for (int r = 0; r < 8; ++r) {
      const int p = mi * 16 + r + 8 * half;
      const int c = nj * 16 + lr;
      const long hw = (long)(row0 + (p >> 3)) * WW + (col0 + (p & 7));
      const float v2 = (1.f - mf) * (float)Vt[c * 64 + p] * sa[bb * HW + hw];
      outsp[(bb * DIMC + c) * (long)HW + hw] = acc[r] + v2;
    }
  }
}

// ================= depthwise 3x3 conv (optionally dilated) =================
__global__ __launch_bounds__(256)
void dwconv_kernel(const float* __restrict__ in, const float* __restrict__ wgt,
                   const float* __restrict__ bias, float* __restrict__ out, int dil) {
  const long idx = (long)blockIdx.x * 256 + threadIdx.x;   // < B*64*HW
  const long hw = idx & (HW - 1);
  const long bc = idx >> 16;
  const int c = (int)(bc & 63);
  const int h = (int)(hw >> 8), wc = (int)(hw & 255);
  float acc = bias[c];
  const float* base = in + bc * (long)HW;
#pragma unroll
  for (int ky = 0; ky < 3; ++ky) {
    const int yy = h + (ky - 1) * dil;
    if (yy < 0 || yy >= HH) continue;
#pragma unroll
    for (int kx = 0; kx < 3; ++kx) {
      const int xx = wc + (kx - 1) * dil;
      if (xx < 0 || xx >= WW) continue;
      acc += wgt[c * 9 + ky * 3 + kx] * base[(long)yy * WW + xx];
    }
  }
  out[idx] = acc;
}

// ====== second depthwise (dil=2) fused with gelu*ca + out_sp ======
__global__ __launch_bounds__(256)
void dwmix_kernel(const float* __restrict__ cs, const float* __restrict__ wgt,
                  const float* __restrict__ bias, const float* __restrict__ outsp,
                  const float* __restrict__ ca, float* __restrict__ out) {
  const long idx = (long)blockIdx.x * 256 + threadIdx.x;
  const long hw = idx & (HW - 1);
  const long bc = idx >> 16;
  const int c = (int)(bc & 63);
  const int h = (int)(hw >> 8), wc = (int)(hw & 255);
  float acc = bias[c];
  const float* base = cs + bc * (long)HW;
#pragma unroll
  for (int ky = 0; ky < 3; ++ky) {
    const int yy = h + (ky - 1) * 2;
    if (yy < 0 || yy >= HH) continue;
#pragma unroll
    for (int kx = 0; kx < 3; ++kx) {
      const int xx = wc + (kx - 1) * 2;
      if (xx < 0 || xx >= WW) continue;
      acc += wgt[c * 9 + ky * 3 + kx] * base[(long)yy * WW + xx];
    }
  }
  const float g = 0.5f * acc * (1.f + erff(acc * 0.70710678118654752f));
  out[idx] = g * ca[bc] + outsp[idx];
}

// =================================================================
extern "C" void kernel_launch(void* const* d_in, const int* in_sizes, int n_in,
                              void* d_out, int out_size, void* d_ws, size_t ws_size,
                              hipStream_t stream) {
  const float* x     = (const float*)d_in[0];
  const float* cg    = (const float*)d_in[1];
  const float* gu    = (const float*)d_in[2];
  const float* pv_w  = (const float*)d_in[3];
  const float* pv_b  = (const float*)d_in[4];
  const float* pq_w  = (const float*)d_in[5];
  const float* pq_b  = (const float*)d_in[6];
  const float* pk_w  = (const float*)d_in[7];
  const float* pk_b  = (const float*)d_in[8];
  const float* cs1_w = (const float*)d_in[9];
  const float* cs1_b = (const float*)d_in[10];
  const float* cs2_w = (const float*)d_in[11];
  const float* cs2_b = (const float*)d_in[12];
  const float* po_w  = (const float*)d_in[13];
  const float* po_b  = (const float*)d_in[14];
  const float* in_w  = (const float*)d_in[15];
  const float* in_b  = (const float*)d_in[16];
  const float* ln_w  = (const float*)d_in[17];
  const float* ln_b  = (const float*)d_in[18];
  const float* o1_w  = (const float*)d_in[19];
  const float* o1_b  = (const float*)d_in[20];
  const float* o2_w  = (const float*)d_in[21];
  const float* o2_b  = (const float*)d_in[22];
  const float* m1_w  = (const float*)d_in[23];
  const float* m1_b  = (const float*)d_in[24];
  const float* m2_w  = (const float*)d_in[25];
  const float* m2_b  = (const float*)d_in[26];
  const float* ca_w  = (const float*)d_in[27];
  const float* ca_b  = (const float*)d_in[28];
  const float* sa_w  = (const float*)d_in[29];
  const float* sa_b  = (const float*)d_in[30];

  // ---- workspace carve (floats) ----
  float* p = (float*)d_ws;
  float* v_buf   = p; p += (size_t)BATCH * DIMC * HW;   // 16.78M
  float* t_buf   = p; p += (size_t)BATCH * C4N * HW;    //  4.46M
  float* tmean_b = p; p += (size_t)BATCH * HW;
  float* offs_b  = p; p += (size_t)BATCH * 2 * HW;
  float* sa_buf  = p; p += (size_t)BATCH * HW;
  float* tsum_b  = p; p += BATCH * C4N;
  float* ca_buf  = p; p += BATCH * DIMC;
  float* mask_b  = p; p += BATCH * NWIN;
  float* kimg_b  = p; p += (size_t)BATCH * DIMC * HW;   // 16.78M
  float* outsp_b = p; p += (size_t)BATCH * DIMC * HW;   // 16.78M
  float* cs_buf  = v_buf;     // v dead after attention
  float* mix_buf = kimg_b;    // k_img dead after attention

  const int pixBlocks = (BATCH * HW) / 256;         // 1024
  const int elemBlocks = (BATCH * DIMC * HW) / 256; // 65536

  // 1. v = pv(x)                          [WMMA]
  conv1x1_wmma<<<(BATCH * HW) / 128, 256, 0, stream>>>(x, pv_w, pv_b, v_buf);
  // 2. predictor: t, tmean, offsets, tsum
  hipMemsetAsync(tsum_b, 0, BATCH * C4N * sizeof(float), stream);
  predictor_kernel<<<pixBlocks, 256, 0, stream>>>(v_buf, cg, in_w, in_b, ln_w, ln_b,
                                                  o1_w, o1_b, o2_w, o2_b,
                                                  t_buf, tmean_b, offs_b, tsum_b);
  // 3. sa / ca / mask
  sa_kernel<<<pixBlocks, 256, 0, stream>>>(t_buf, sa_w, sa_b, sa_buf);
  ca_kernel<<<BATCH, DIMC, 0, stream>>>(tsum_b, ca_w, ca_b, ca_buf);
  mask_kernel<<<(BATCH * NWIN) / 256, 256, 0, stream>>>(tmean_b, gu, m1_w, m1_b,
                                                        m2_w, m2_b, mask_b);
  // 4. k_img = x + warp(x, offsets)
  warp_kernel<<<pixBlocks, 256, 0, stream>>>(x, offs_b, kimg_b);
  // 5. windowed attention                 [WMMA]
  attn_wmma<<<BATCH * NWIN, 256, 0, stream>>>(x, kimg_b, v_buf, sa_buf, mask_b,
                                              pq_w, pq_b, pk_w, pk_b, outsp_b);
  // 6. depthwise convs + gelu*ca + residual
  dwconv_kernel<<<elemBlocks, 256, 0, stream>>>(outsp_b, cs1_w, cs1_b, cs_buf, 1);
  dwmix_kernel<<<elemBlocks, 256, 0, stream>>>(cs_buf, cs2_w, cs2_b, outsp_b,
                                               ca_buf, mix_buf);
  // 7. final projection                   [WMMA]
  conv1x1_wmma<<<(BATCH * HW) / 128, 256, 0, stream>>>(mix_buf, po_w, po_b,
                                                       (float*)d_out);
}
